// MLA_57629871177816
// MI455X (gfx1250) — matrix operand order repaired and checked by reading, not compiled, not run on hardware
//
#include <hip/hip_runtime.h>

// ---------------------------------------------------------------------------
// Types / helpers
// ---------------------------------------------------------------------------
typedef __bf16 bf16;
typedef __attribute__((ext_vector_type(16))) __bf16     v16bf;
typedef __attribute__((ext_vector_type(8)))  float      v8f;
typedef __attribute__((ext_vector_type(4)))  unsigned   u32x4;
typedef __attribute__((ext_vector_type(2)))  unsigned   u32x2;
typedef __attribute__((ext_vector_type(4)))  float      f32x4;

union FragAB { v16bf v; u32x4 u[2]; };

__device__ inline bf16 f2bf(float f) {
  unsigned u = __builtin_bit_cast(unsigned, f);
  u += 0x7fffu + ((u >> 16) & 1u);   // round-to-nearest-even
  unsigned short h = (unsigned short)(u >> 16);
  return __builtin_bit_cast(bf16, h);
}
__device__ inline float bf2f(unsigned short h) {
  unsigned u = ((unsigned)h) << 16;
  return __builtin_bit_cast(float, u);
}
__device__ inline unsigned pack2bf(float a, float b) {
  unsigned lo = (unsigned)__builtin_bit_cast(unsigned short, f2bf(a));
  unsigned hi = (unsigned)__builtin_bit_cast(unsigned short, f2bf(b));
  return lo | (hi << 16);
}
__device__ inline v8f v8f_zero() {
  v8f v;
  #pragma unroll
  for (int i = 0; i < 8; ++i) v[i] = 0.0f;
  return v;
}

// CDNA5 async global->LDS copy (16 bytes per lane), tracked by ASYNCcnt.
__device__ inline void async_copy16(unsigned lds_off, const void* gaddr) {
  asm volatile("global_load_async_to_lds_b128 %0, %1, off"
               :: "v"(lds_off), "v"(gaddr) : "memory");
}
__device__ inline void async_wait0() {
  asm volatile("s_wait_asynccnt 0x0" ::: "memory");
}
__device__ inline unsigned lds_off(const void* p) {
  return (unsigned)(unsigned long long)p;   // low 32 bits = LDS byte offset
}

// ---------------------------------------------------------------------------
// One-shot conversions
// ---------------------------------------------------------------------------
__global__ __launch_bounds__(256)
void cvt_f32_to_bf16(const float* __restrict__ in, bf16* __restrict__ out, long n4)
{
  long i = (long)blockIdx.x * blockDim.x + threadIdx.x;
  if (i >= n4) return;
  f32x4 v = *(const f32x4*)(in + i * 4);
  u32x2 p; p.x = pack2bf(v.x, v.y); p.y = pack2bf(v.z, v.w);
  *(u32x2*)(out + i * 4) = p;
}

// Bt[n*K + k] = bf16(B[k*N + n])
__global__ __launch_bounds__(256)
void transpose_cvt(const float* __restrict__ B, bf16* __restrict__ Bt, int K, int N)
{
  __shared__ float tile[32][33];
  const int k0 = blockIdx.y * 32, n0 = blockIdx.x * 32;
  const int tx = threadIdx.x & 31, ty = threadIdx.x >> 5;   // 32 x 8
  #pragma unroll
  for (int i = 0; i < 4; ++i) {
    int k = k0 + ty + i * 8, n = n0 + tx;
    tile[ty + i * 8][tx] = (k < K && n < N) ? B[(long)k * N + n] : 0.0f;
  }
  __syncthreads();
  #pragma unroll
  for (int i = 0; i < 4; ++i) {
    int n = n0 + ty + i * 8, k = k0 + tx;
    if (n < N && k < K) Bt[(long)n * K + k] = f2bf(tile[tx][ty + i * 8]);
  }
}

// ---------------------------------------------------------------------------
// GEMM: C[M,N] = A[M,K](bf16, lda) * Bt[N,K](bf16, K-major) + bias[N]
// Block tile 256x128, K-step 64, 256 threads = 8 waves (4x2), each wave 64x64
// (4x4 WMMA tiles -> 1.0 ds_load per wmma). Double-buffered async staging.
// ---------------------------------------------------------------------------
#define BM 256
#define BN 128
#define BKK 64

__global__ __launch_bounds__(256)
void gemm_bf16_wmma(const bf16* __restrict__ A,  int lda,
                    const bf16* __restrict__ Bt, int ldb,
                    int K,
                    float* __restrict__ Cf, bf16* __restrict__ Cb, int ldc,
                    const float* __restrict__ bias,
                    int M, int N)
{
  __shared__ bf16 As[2][BM][BKK];   // 2 x 32 KB
  __shared__ bf16 Bs[2][BN][BKK];   // 2 x 16 KB

  const int tid  = threadIdx.x;
  const int lane = tid & 31;
  const int wid  = tid >> 5;
  const int wm   = (wid & 3) * 64;
  const int wn   = (wid >> 2) * 64;
  const int bm   = blockIdx.y * BM;
  const int bn   = blockIdx.x * BN;

  v8f acc[4][4];
  #pragma unroll
  for (int mi = 0; mi < 4; ++mi)
    #pragma unroll
    for (int ni = 0; ni < 4; ++ni) acc[mi][ni] = v8f_zero();

  // staging: A = 2048 chunks of 16B (8/thread), B = 1024 chunks (4/thread)
  auto stage = [&](int buf, int k0) {
    const char* Ab = (const char*)A  + ((size_t)bm * lda + k0) * 2;
    const char* Bb = (const char*)Bt + ((size_t)bn * ldb + k0) * 2;
    #pragma unroll
    for (int i = 0; i < 8; ++i) {
      int c = tid + i * 256;
      int row = c >> 3, off = (c & 7) * 16;
      if (bm + row < M)
        async_copy16(lds_off(&As[buf][row][0]) + off, Ab + (size_t)row * lda * 2 + off);
    }
    #pragma unroll
    for (int i = 0; i < 4; ++i) {
      int c = tid + i * 256;
      int row = c >> 3, off = (c & 7) * 16;
      if (bn + row < N)
        async_copy16(lds_off(&Bs[buf][row][0]) + off, Bb + (size_t)row * ldb * 2 + off);
    }
  };

  stage(0, 0);
  async_wait0();
  __syncthreads();

  int buf = 0;
  for (int k0 = 0; k0 < K; k0 += BKK) {
    if (k0 + BKK < K) stage(buf ^ 1, k0 + BKK);

    #pragma unroll
    for (int kk = 0; kk < BKK; kk += 32) {
      FragAB af[4], bfr[4];
      #pragma unroll
      for (int mi = 0; mi < 4; ++mi) {
        const bf16* p = &As[buf][wm + mi * 16 + (lane & 15)][kk + ((lane >> 4) * 8)];
        af[mi].u[0] = *(const u32x4*)(p);
        af[mi].u[1] = *(const u32x4*)(p + 16);
      }
      #pragma unroll
      for (int ni = 0; ni < 4; ++ni) {
        const bf16* p = &Bs[buf][wn + ni * 16 + (lane & 15)][kk + ((lane >> 4) * 8)];
        bfr[ni].u[0] = *(const u32x4*)(p);
        bfr[ni].u[1] = *(const u32x4*)(p + 16);
      }
      #pragma unroll
      for (int mi = 0; mi < 4; ++mi)
        #pragma unroll
        for (int ni = 0; ni < 4; ++ni)
          acc[mi][ni] = __builtin_amdgcn_wmma_f32_16x16x32_bf16(
              false, af[mi].v, false, bfr[ni].v, (short)0, acc[mi][ni],
              false, false);
    }
    async_wait0();
    __syncthreads();
    buf ^= 1;
  }

  const int rb = (lane >> 4) * 8;
  const int cl = lane & 15;
  #pragma unroll
  for (int mi = 0; mi < 4; ++mi) {
    #pragma unroll
    for (int ni = 0; ni < 4; ++ni) {
      int col = bn + wn + ni * 16 + cl;
      if (col >= N) continue;
      float bv = bias ? bias[col] : 0.0f;
      #pragma unroll
      for (int i = 0; i < 8; ++i) {
        int row = bm + wm + mi * 16 + rb + i;
        if (row >= M) continue;
        float v = acc[mi][ni][i] + bv;
        if (Cf) Cf[(long)row * ldc + col] = v;
        if (Cb) Cb[(long)row * ldc + col] = f2bf(v);
      }
    }
  }
}

// ---------------------------------------------------------------------------
// Interleaved RoPE, in place on bf16: x layout [rows][H*64], t = row % T
// ---------------------------------------------------------------------------
__global__ void rope_bf16(bf16* __restrict__ x, int rows, int T, int H)
{
  int idx   = blockIdx.x * blockDim.x + threadIdx.x;
  int total = rows * H * 32;
  if (idx >= total) return;
  int j   = idx & 31;
  int h   = (idx >> 5) % H;
  int row = idx / (32 * H);
  int t   = row % T;
  float theta = __powf(10000.0f, -2.0f * (float)j / 64.0f);
  float s, c;
  __sincosf((float)t * theta, &s, &c);
  unsigned* p = (unsigned*)(x + (long)row * (H * 64) + h * 64 + 2 * j);
  unsigned w = *p;
  float x0 = bf2f((unsigned short)(w & 0xffff));
  float x1 = bf2f((unsigned short)(w >> 16));
  *p = pack2bf(x0 * c - x1 * s, x1 * c + x0 * s);
}

// ---------------------------------------------------------------------------
// Flash attention (MLA). One block = (b,h) x 128 query rows, 256 threads.
// TQ=128 halves K/V staging per FLOP; waves: S -> 4x2 of 32x32 (4 wmma/step),
// O -> 4x2 of 32x64 (8 wmma/step). Async staging for Q/K, manual V transpose.
// ---------------------------------------------------------------------------
#define TQ   128
#define TK   64
#define ADIM 192
#define DKV  128

__global__ __launch_bounds__(256)
void mla_attn(const bf16* __restrict__ qu,   // [B*T, 2048]
              const bf16* __restrict__ qr,   // [B*T, 1024]
              const bf16* __restrict__ kv,   // [B*T, 4096] (v | k per head)
              const bf16* __restrict__ kr,   // [B*T, 64]
              const float* __restrict__ amask,// [B*T]
              bf16* __restrict__ out,        // [B*T, 2048] col = h*128+d
              int T)
{
  __shared__ bf16  Qs[TQ][ADIM];   // 48 KB
  __shared__ bf16  Ks[TK][ADIM];   // 24 KB
  __shared__ bf16  Vt[DKV][TK];    // 16 KB
  __shared__ float Ss[TQ][TK];     // 32 KB
  __shared__ bf16  Ps[TQ][TK];     // 16 KB
  __shared__ float mrow[TQ], lrow[TQ], arow[TQ];

  const int tid  = threadIdx.x;
  const int lane = tid & 31;
  const int wid  = tid >> 5;
  const int bh   = blockIdx.y;
  const int b    = bh >> 4;
  const int h    = bh & 15;
  const int q0   = blockIdx.x * TQ;
  const float scale = 0.08838834764831845f;  // 1/sqrt(128)

  // ---- stage Q block via async copies: 128 rows x (16 + 8) 16B chunks -----
  #pragma unroll
  for (int i = 0; i < 12; ++i) {
    int c = tid + i * 256;
    int r = c / 24, j = c % 24;
    long grow = (long)(b * T + q0 + r);
    unsigned l = lds_off(&Qs[r][0]) + ((j < 16) ? j * 16 : 256 + (j - 16) * 16);
    const char* g = (j < 16)
        ? (const char*)qu + ((size_t)grow * 2048 + h * 128) * 2 + j * 16
        : (const char*)qr + ((size_t)grow * 1024 + h * 64) * 2 + (j - 16) * 16;
    async_copy16(l, g);
  }
  if (tid < TQ) { mrow[tid] = -3.0e38f; lrow[tid] = 0.0f; }

  const int s_wm = (wid & 3) * 32;     // S: 4x2 waves of 32x32
  const int s_wn = (wid >> 2) * 32;
  const int o_wm = (wid & 3) * 32;     // O: 4x2 waves of 32x64
  const int o_wn = (wid >> 2) * 64;

  v8f oacc[2][4];
  #pragma unroll
  for (int i = 0; i < 2; ++i)
    #pragma unroll
    for (int j = 0; j < 4; ++j) oacc[i][j] = v8f_zero();

  async_wait0();
  __syncthreads();

  const int nkb = 2 * (blockIdx.x + 1);   // causal: keys 0 .. q0+TQ-1
  for (int kb = 0; kb < nkb; ++kb) {
    const int k0 = kb * TK;

    // ---- stage K block (async, 64 rows x 24 chunks) -----------------------
    #pragma unroll
    for (int i = 0; i < 6; ++i) {
      int c = tid + i * 256;
      int r = c / 24, j = c % 24;
      long grow = (long)(b * T + k0 + r);
      unsigned l = lds_off(&Ks[r][0]) + ((j < 16) ? j * 16 : 256 + (j - 16) * 16);
      const char* g = (j < 16)
          ? (const char*)kv + ((size_t)grow * 4096 + h * 256 + 128) * 2 + j * 16
          : (const char*)kr + ((size_t)grow * 64) * 2 + (j - 16) * 16;
      async_copy16(l, g);
    }
    // ---- stage V transposed (manual) --------------------------------------
    #pragma unroll 4
    for (int e = tid; e < TK * DKV / 2; e += 256) {
      int r = e >> 6;            // key row
      int d = (e & 63) * 2;
      unsigned w = *(const unsigned*)((const char*)kv +
                    ((size_t)(b * T + k0 + r) * 4096 + h * 256 + d) * 2);
      Vt[d][r]     = __builtin_bit_cast(bf16, (unsigned short)(w & 0xffff));
      Vt[d + 1][r] = __builtin_bit_cast(bf16, (unsigned short)(w >> 16));
    }
    async_wait0();
    __syncthreads();

    // ---- S = Q K^T (each wave 32x32, 6 K-steps) ---------------------------
    v8f sacc[2][2];
    #pragma unroll
    for (int mi = 0; mi < 2; ++mi)
      #pragma unroll
      for (int ni = 0; ni < 2; ++ni) sacc[mi][ni] = v8f_zero();
    #pragma unroll
    for (int kk = 0; kk < ADIM; kk += 32) {
      FragAB a[2], bb[2];
      #pragma unroll
      for (int mi = 0; mi < 2; ++mi) {
        const bf16* p = &Qs[s_wm + mi * 16 + (lane & 15)][kk + ((lane >> 4) * 8)];
        a[mi].u[0] = *(const u32x4*)(p); a[mi].u[1] = *(const u32x4*)(p + 16);
      }
      #pragma unroll
      for (int ni = 0; ni < 2; ++ni) {
        const bf16* p = &Ks[s_wn + ni * 16 + (lane & 15)][kk + ((lane >> 4) * 8)];
        bb[ni].u[0] = *(const u32x4*)(p); bb[ni].u[1] = *(const u32x4*)(p + 16);
      }
      #pragma unroll
      for (int mi = 0; mi < 2; ++mi)
        #pragma unroll
        for (int ni = 0; ni < 2; ++ni)
          sacc[mi][ni] = __builtin_amdgcn_wmma_f32_16x16x32_bf16(
              false, a[mi].v, false, bb[ni].v, (short)0, sacc[mi][ni],
              false, false);
    }
    // ---- scale + mask + spill to LDS --------------------------------------
    {
      const int rb = (lane >> 4) * 8;
      const int cl = lane & 15;
      #pragma unroll
      for (int mi = 0; mi < 2; ++mi) {
        #pragma unroll
        for (int ni = 0; ni < 2; ++ni) {
          #pragma unroll
          for (int i = 0; i < 8; ++i) {
            int r = s_wm + mi * 16 + rb + i;
            int c = s_wn + ni * 16 + cl;
            int qrow = q0 + r, krow = k0 + c;
            float v = sacc[mi][ni][i] * scale;
            if (krow > qrow) v = -9.0e15f;
            if (amask[(long)b * T + krow] == 0.0f) v = -9.0e15f;
            Ss[r][c] = v;
          }
        }
      }
    }
    __syncthreads();

    // ---- online softmax (one thread per row) ------------------------------
    if (tid < TQ) {
      int r = tid;
      float m_old = mrow[r];
      float mx = m_old;
      #pragma unroll 8
      for (int c = 0; c < TK; ++c) mx = fmaxf(mx, Ss[r][c]);
      float alpha = __expf(m_old - mx);
      float sum = 0.0f;
      #pragma unroll 4
      for (int c = 0; c < TK; ++c) {
        float p = __expf(Ss[r][c] - mx);
        sum += p;
        Ps[r][c] = f2bf(p);
      }
      mrow[r] = mx;
      lrow[r] = lrow[r] * alpha + sum;
      arow[r] = alpha;
    }
    __syncthreads();

    // ---- rescale O then O += P V ------------------------------------------
    {
      const int rb = (lane >> 4) * 8;
      #pragma unroll
      for (int mi = 0; mi < 2; ++mi) {
        #pragma unroll
        for (int i = 0; i < 8; ++i) {
          float al = arow[o_wm + mi * 16 + rb + i];
          #pragma unroll
          for (int ni = 0; ni < 4; ++ni) oacc[mi][ni][i] *= al;
        }
      }
      #pragma unroll
      for (int kk = 0; kk < TK; kk += 32) {
        FragAB a[2], bb[4];
        #pragma unroll
        for (int mi = 0; mi < 2; ++mi) {
          const bf16* p = &Ps[o_wm + mi * 16 + (lane & 15)][kk + ((lane >> 4) * 8)];
          a[mi].u[0] = *(const u32x4*)(p); a[mi].u[1] = *(const u32x4*)(p + 16);
        }
        #pragma unroll
        for (int ni = 0; ni < 4; ++ni) {
          const bf16* p = &Vt[o_wn + ni * 16 + (lane & 15)][kk + ((lane >> 4) * 8)];
          bb[ni].u[0] = *(const u32x4*)(p); bb[ni].u[1] = *(const u32x4*)(p + 16);
        }
        #pragma unroll
        for (int mi = 0; mi < 2; ++mi)
          #pragma unroll
          for (int ni = 0; ni < 4; ++ni)
            oacc[mi][ni] = __builtin_amdgcn_wmma_f32_16x16x32_bf16(
                false, a[mi].v, false, bb[ni].v, (short)0, oacc[mi][ni],
                false, false);
      }
    }
    __syncthreads();
  }

  // ---- normalize and write out (bf16) -------------------------------------
  {
    const int rb = (lane >> 4) * 8;
    const int cl = lane & 15;
    #pragma unroll
    for (int mi = 0; mi < 2; ++mi) {
      #pragma unroll
      for (int ni = 0; ni < 4; ++ni) {
        int c = o_wn + ni * 16 + cl;
        #pragma unroll
        for (int i = 0; i < 8; ++i) {
          int r = o_wm + mi * 16 + rb + i;
          float inv = 1.0f / lrow[r];
          out[(size_t)(b * T + q0 + r) * 2048 + h * DKV + c] =
              f2bf(oacc[mi][ni][i] * inv);
        }
      }
    }
  }
}

// ---------------------------------------------------------------------------
// Host-side orchestration
// ---------------------------------------------------------------------------
extern "C" void kernel_launch(void* const* d_in, const int* in_sizes, int n_in,
                              void* d_out, int out_size, void* d_ws, size_t ws_size,
                              hipStream_t stream)
{
  (void)in_sizes; (void)n_in; (void)out_size; (void)ws_size;
  const float* x    = (const float*)d_in[0];
  const float* am   = (const float*)d_in[1];
  const float* Wd   = (const float*)d_in[2];
  const float* Wdb  = (const float*)d_in[3];
  const float* Wkv  = (const float*)d_in[4];
  const float* Wkvb = (const float*)d_in[5];
  const float* Wq   = (const float*)d_in[6];
  const float* Wqb  = (const float*)d_in[7];
  const float* Wkr  = (const float*)d_in[8];
  const float* Wkrb = (const float*)d_in[9];
  const float* Wqr  = (const float*)d_in[10];
  const float* Wqrb = (const float*)d_in[11];
  const float* Wo   = (const float*)d_in[12];
  const float* Wob  = (const float*)d_in[13];
  float* out = (float*)d_out;

  const int B = 2, T = 2048, C = 2048, NH = 16, L = 512;
  const int M = B * T;  // 4096

  bf16* w = (bf16*)d_ws;
  bf16* xb   = w;  w += (size_t)M * 2048;
  bf16* dnb  = w;  w += (size_t)M * 1024;    // [cKV | cq]
  bf16* krb  = w;  w += (size_t)M * 64;
  bf16* qrb  = w;  w += (size_t)M * 1024;
  bf16* kvb  = w;  w += (size_t)M * 4096;    // per head: [v | k]
  bf16* qub  = w;  w += (size_t)M * 2048;
  bf16* aob  = w;  w += (size_t)M * 2048;
  bf16* WdT  = w;  w += (size_t)1024 * 2048;
  bf16* WkrT = w;  w += (size_t)64 * 2048;
  bf16* WqrT = w;  w += (size_t)1024 * 2048;
  bf16* WkvT = w;  w += (size_t)4096 * 512;
  bf16* WqT  = w;  w += (size_t)2048 * 512;
  bf16* WoT  = w;

  dim3 blk(256);
  auto grd  = [](int n, int m) { return dim3((n + BN - 1) / BN, (m + BM - 1) / BM); };
  auto tgrd = [](int K, int N) { return dim3((N + 31) / 32, (K + 31) / 32); };

  // one-shot operand conversion / weight transposition
  {
    long n4 = (long)M * 2048 / 4;
    cvt_f32_to_bf16<<<dim3((unsigned)((n4 + 255) / 256)), blk, 0, stream>>>(x, xb, n4);
  }
  transpose_cvt<<<tgrd(2048, 1024), blk, 0, stream>>>(Wd,  WdT,  2048, 1024);
  transpose_cvt<<<tgrd(2048, 64),   blk, 0, stream>>>(Wkr, WkrT, 2048, 64);
  transpose_cvt<<<tgrd(2048, 1024), blk, 0, stream>>>(Wqr, WqrT, 2048, 1024);
  transpose_cvt<<<tgrd(512, 4096),  blk, 0, stream>>>(Wkv, WkvT, 512, 4096);
  transpose_cvt<<<tgrd(512, 2048),  blk, 0, stream>>>(Wq,  WqT,  512, 2048);
  transpose_cvt<<<tgrd(2048, 2048), blk, 0, stream>>>(Wo,  WoT,  2048, 2048);

  // down / rope projections (K = 2048), bf16 outputs
  gemm_bf16_wmma<<<grd(1024, M), blk, 0, stream>>>(xb, C, WdT,  C, C, nullptr, dnb, 1024, Wdb,  M, 1024);
  gemm_bf16_wmma<<<grd(64,   M), blk, 0, stream>>>(xb, C, WkrT, C, C, nullptr, krb, 64,   Wkrb, M, 64);
  gemm_bf16_wmma<<<grd(1024, M), blk, 0, stream>>>(xb, C, WqrT, C, C, nullptr, qrb, 1024, Wqrb, M, 1024);

  // RoPE (in place, bf16)
  {
    int total = M * 1 * 32;
    rope_bf16<<<dim3((total + 255) / 256), blk, 0, stream>>>(krb, M, T, 1);
  }
  {
    int total = M * 16 * 32;
    rope_bf16<<<dim3((total + 255) / 256), blk, 0, stream>>>(qrb, M, T, 16);
  }

  // up projections (K = 512); cKV = dnb[:, :512], cq = dnb[:, 512:]
  gemm_bf16_wmma<<<grd(4096, M), blk, 0, stream>>>(dnb,       1024, WkvT, L, L, nullptr, kvb, 4096, Wkvb, M, 4096);
  gemm_bf16_wmma<<<grd(2048, M), blk, 0, stream>>>(dnb + 512, 1024, WqT,  L, L, nullptr, qub, 2048, Wqb,  M, 2048);

  // flash attention: grid = (T/128 query blocks, B*NH)
  mla_attn<<<dim3(T / TQ, B * NH), blk, 0, stream>>>(qub, qrb, kvb, krb, am, aob, T);

  // output projection -> f32 result
  gemm_bf16_wmma<<<grd(2048, M), blk, 0, stream>>>(aob, 2048, WoT, C, C, out, nullptr, 2048, Wob, M, 2048);
}
